// SelfAttention_11132555231233
// MI455X (gfx1250) — matrix operand order repaired and checked by reading, not compiled, not run on hardware
//
#include <hip/hip_runtime.h>
#include <hip/hip_bf16.h>

// ---------------------------------------------------------------------------
// Self-attention forward for MI455X (gfx1250), wave32 + v_wmma_f32_16x16x32_bf16
//   N=8, S=1024 tokens, E=1024, HEADS=16, HEAD_DIM=64
// (1) per-head QKV projection f32->bf16 (WMMA)
// (2) flash attention: K/V tiles staged in LDS once per block (4 waves share),
//     WMMA scores + log2-domain online softmax + WMMA P*V, 64 keys/step
// (3) 8192x1024x1024 output GEMM + bias (WMMA, 64x64 tile/wave)
// Intermediates (~69MB bf16) live in d_ws and stay resident in the 192MB L2.
// All WMMA operand tiles are fetched as b128 loads (per-lane-contiguous runs).
// ---------------------------------------------------------------------------

#define NB    8
#define HEADS 16
#define SEQ   1024
#define HD    64
#define EMB   1024
#define LDP   72          // padded LDS row stride (bank-conflict-free: 36 banks)

typedef __attribute__((ext_vector_type(16))) __bf16 v16bf;
typedef __attribute__((ext_vector_type(8)))  float  v8f;

struct U8 { unsigned u[8]; };

static __device__ __forceinline__ unsigned short f2bf_raw(float f) {
  unsigned u = __builtin_bit_cast(unsigned, f);
  return (unsigned short)((u + 0x8000u) >> 16);   // round-to-nearest
}
// pack two f32 -> two bf16 in one dword: 2x v_add + 1x v_perm_b32
static __device__ __forceinline__ unsigned packbf2(float lo, float hi) {
  unsigned a = __builtin_bit_cast(unsigned, lo) + 0x8000u;
  unsigned b = __builtin_bit_cast(unsigned, hi) + 0x8000u;
#if __has_builtin(__builtin_amdgcn_perm)
  return __builtin_amdgcn_perm(b, a, 0x07060302u);   // {b[31:16], a[31:16]}
#else
  return (a >> 16) | (b & 0xffff0000u);
#endif
}

// A tile (16 rows x K=32) from bf16 row-major [row][ld].
// ISA 16-bit A layout: lane m=l&15, group g=l>>4; VGPR j holds K pair
// k0 = (j&3)*2 + 8g + 16*(j>>2)  -> per lane: elements [8g,8g+7] and
// [16+8g,16+8g+7] are contiguous: two b128 loads.
static __device__ __forceinline__ v16bf loadA_bf16(const unsigned short* base,
                                                   int ld, int kb, int lane) {
  int m = lane & 15, g = lane >> 4;
  const unsigned short* p = base + (size_t)m * ld + kb + 8 * g;
  uint4 lo = *(const uint4*)p;          // k = 8g .. 8g+7
  uint4 hi = *(const uint4*)(p + 16);   // k = 16+8g .. 16+8g+7
  U8 t = {{lo.x, lo.y, lo.z, lo.w, hi.x, hi.y, hi.z, hi.w}};
  return __builtin_bit_cast(v16bf, t);
}

static __device__ __forceinline__ v16bf loadA_f32(const float* base,
                                                  int ld, int kb, int lane) {
  int m = lane & 15, g = lane >> 4;
  const float* p = base + (size_t)m * ld + kb + 8 * g;
  float4 f0 = *(const float4*)p;
  float4 f1 = *(const float4*)(p + 4);
  float4 f2 = *(const float4*)(p + 16);
  float4 f3 = *(const float4*)(p + 20);
  U8 t = {{packbf2(f0.x, f0.y), packbf2(f0.z, f0.w),
           packbf2(f1.x, f1.y), packbf2(f1.z, f1.w),
           packbf2(f2.x, f2.y), packbf2(f2.z, f2.w),
           packbf2(f3.x, f3.y), packbf2(f3.z, f3.w)}};
  return __builtin_bit_cast(v16bf, t);
}

// B tile (K=32 x N=16), element B[k][n] = src[n*ld + kb + k].
// ISA 16-bit B layout: lane n=l&15, g=l>>4; same K order as A:
// -> per lane: elements [8g,8g+7] and [16+8g,16+8g+7]: two b128 loads.
static __device__ __forceinline__ v16bf loadB_bf16(const unsigned short* src,
                                                   int ld, int kb, int lane) {
  int n = lane & 15, g = lane >> 4;
  const unsigned short* p = src + (size_t)n * ld + kb + 8 * g;
  uint4 a = *(const uint4*)p;           // k = 8g .. 8g+7
  uint4 b = *(const uint4*)(p + 16);    // k = 16+8g .. 16+8g+7
  U8 t = {{a.x, a.y, a.z, a.w, b.x, b.y, b.z, b.w}};
  return __builtin_bit_cast(v16bf, t);
}

static __device__ __forceinline__ v16bf loadB_f32(const float* src,
                                                  int ld, int kb, int lane) {
  int n = lane & 15, g = lane >> 4;
  const float* p = src + (size_t)n * ld + kb + 8 * g;
  float4 f0 = *(const float4*)p;
  float4 f1 = *(const float4*)(p + 4);
  float4 f2 = *(const float4*)(p + 16);
  float4 f3 = *(const float4*)(p + 20);
  U8 t = {{packbf2(f0.x, f0.y), packbf2(f0.z, f0.w),
           packbf2(f1.x, f1.y), packbf2(f1.z, f1.w),
           packbf2(f2.x, f2.y), packbf2(f2.z, f2.w),
           packbf2(f3.x, f3.y), packbf2(f3.z, f3.w)}};
  return __builtin_bit_cast(v16bf, t);
}

static __device__ __forceinline__ v8f wmma_bf16(v16bf a, v16bf b, v8f c) {
  return __builtin_amdgcn_wmma_f32_16x16x32_bf16(false, a, false, b,
                                                 (short)0, c, false, false);
}

// ---------------------------------------------------------------------------
// Kernel 0: Wo f32 -> bf16 (read once, reused from L2 by oproj)
// ---------------------------------------------------------------------------
__global__ __launch_bounds__(256) void cvt_wo_kernel(const float* __restrict__ Wo,
                                                     unsigned short* __restrict__ wob) {
  int i = (blockIdx.x * 256 + threadIdx.x) * 4;   // EMB*EMB = 1M elements
  float4 f = *(const float4*)(Wo + i);
  uint2 st;
  st.x = packbf2(f.x, f.y);
  st.y = packbf2(f.z, f.w);
  *(uint2*)(wob + i) = st;
}

// ---------------------------------------------------------------------------
// Kernel 1: per-head projections  y[.,e] = sum_d x[.,d] * W[e,d]
// grid: (token_tiles=512, heads=16, which=3), 1 wave/block.
// Q',K' stored [n*H+h][s][d] bf16; V' stored transposed [n*H+h][d][s] bf16.
// ---------------------------------------------------------------------------
__global__ __launch_bounds__(32) void proj_kernel(
    const float* __restrict__ q_in, const float* __restrict__ k_in,
    const float* __restrict__ v_in,
    const float* __restrict__ Wq, const float* __restrict__ Wk,
    const float* __restrict__ Wv,
    unsigned short* __restrict__ qp, unsigned short* __restrict__ kp,
    unsigned short* __restrict__ vp)
{
  int lane = threadIdx.x & 31;
  int t0 = blockIdx.x * 16;            // token = n*SEQ + s  (tile never crosses n)
  int h  = blockIdx.y;
  int which = blockIdx.z;              // 0=q, 1=k, 2=v

  const float* x = (which == 0) ? q_in : (which == 1) ? k_in : v_in;
  const float* W = (which == 0) ? Wq   : (which == 1) ? Wk   : Wv;

  const float* abase = x + ((size_t)t0 * HEADS + h) * HD;   // row stride = HEADS*HD
  v16bf A0 = loadA_f32(abase, HEADS * HD, 0,  lane);
  v16bf A1 = loadA_f32(abase, HEADS * HD, 32, lane);

  int n = t0 / SEQ, s0 = t0 % SEQ;
  int g = lane >> 4, nn = lane & 15;

#pragma unroll
  for (int nt = 0; nt < 4; ++nt) {
    v8f acc = {};
    v16bf B0 = loadB_f32(W + nt * 16 * HD, HD, 0, lane);   // B[d][e] = W[e][d]
    acc = wmma_bf16(A0, B0, acc);
    v16bf B1 = loadB_f32(W + nt * 16 * HD, HD, 32, lane);
    acc = wmma_bf16(A1, B1, acc);

    if (which == 2) {
      // transposed store: vp[(nh*HD + d)*SEQ + s]; 8 consecutive s per lane
      unsigned short* dst =
          vp + ((size_t)(n * HEADS + h) * HD + nt * 16 + nn) * SEQ + s0 + 8 * g;
      uint4 st;
      st.x = packbf2(acc[0], acc[1]);
      st.y = packbf2(acc[2], acc[3]);
      st.z = packbf2(acc[4], acc[5]);
      st.w = packbf2(acc[6], acc[7]);
      *(uint4*)dst = st;
    } else {
      unsigned short* o = (which == 0) ? qp : kp;
      unsigned short* dst =
          o + ((size_t)(n * HEADS + h) * SEQ + s0) * HD + nt * 16 + nn;
#pragma unroll
      for (int r = 0; r < 8; ++r)
        dst[(size_t)(r + 8 * g) * HD] = f2bf_raw(acc[r]);
    }
  }
}

// ---------------------------------------------------------------------------
// Kernel 2: flash attention per (n,h). 1 wave = 16 query rows; 4 waves/block.
// K tile (64x64) and V tile (64x64) staged in LDS once per 64-key step and
// shared by all 4 waves (4x less L2 traffic). LDS rows padded to 72 elements
// so row-strided B reads are bank-conflict-free.
// ---------------------------------------------------------------------------
__global__ __launch_bounds__(128) void attn_kernel(
    const unsigned short* __restrict__ qp, const unsigned short* __restrict__ kp,
    const unsigned short* __restrict__ vp, unsigned short* __restrict__ ao)
{
  __shared__ unsigned short ksh[64 * LDP];          // [key][d]
  __shared__ unsigned short vsh[64 * LDP];          // [d][key]
  __shared__ unsigned short pshare[4 * 16 * LDP];   // P, 16x64 per wave

  int tid = threadIdx.x;
  int lane = tid & 31, wave = tid >> 5;
  int nh = blockIdx.x;                 // n*HEADS + h
  int n = nh / HEADS, h = nh % HEADS;
  int q0 = (blockIdx.y * 4 + wave) * 16;

  const unsigned short* qbase = qp + ((size_t)nh * SEQ + q0) * HD;
  const unsigned short* kbase = kp + (size_t)nh * SEQ * HD;
  const unsigned short* vbase = vp + (size_t)nh * HD * SEQ;   // [d][s]

  v16bf Aq0 = loadA_bf16(qbase, HD, 0,  lane);
  v16bf Aq1 = loadA_bf16(qbase, HD, 32, lane);

  v8f acc[4] = {};
  float mrow[8], lrow[8];
#pragma unroll
  for (int r = 0; r < 8; ++r) { mrow[r] = -1e30f; lrow[r] = 0.0f; }

  unsigned short* pw = pshare + wave * (16 * LDP);
  int g = lane >> 4, nn = lane & 15;
  // scores kept in log2 domain: C = log2(e)/sqrt(EMB)
  const float C = 0.045084220f;

  int crow  = tid >> 1;          // staging: row 0..63
  int chalf = (tid & 1) * 32;    // first/second 32 elements of the row

  for (int kb = 0; kb < SEQ; kb += 64) {
    __syncthreads();   // previous step's LDS reads complete
    {
      // stage K tile [64 keys][64 d] and V tile [64 d][64 keys] (b128 copies)
      const uint4* ks = (const uint4*)(kbase + (size_t)(kb + crow) * HD + chalf);
      uint4* kd = (uint4*)(ksh + crow * LDP + chalf);
      const uint4* vs = (const uint4*)(vbase + (size_t)crow * SEQ + kb + chalf);
      uint4* vd = (uint4*)(vsh + crow * LDP + chalf);
#pragma unroll
      for (int i = 0; i < 4; ++i) { kd[i] = ks[i]; vd[i] = vs[i]; }
    }
    // prefetch next K block (speculative; OOB prefetch is dropped)
    __builtin_prefetch(kbase + (size_t)(kb + 64 + crow) * HD + chalf, 0, 1);
    __syncthreads();   // staged tiles visible to all waves

    // scores: four 16x16 tiles, K=64 (2 WMMA each), B from LDS
    v8f s[4] = {};
#pragma unroll
    for (int t = 0; t < 4; ++t) {
      const unsigned short* kt = ksh + (16 * t) * LDP;
      s[t] = wmma_bf16(Aq0, loadB_bf16(kt, LDP, 0,  lane), s[t]);
      s[t] = wmma_bf16(Aq1, loadB_bf16(kt, LDP, 32, lane), s[t]);
    }

    // online softmax; row M=r+8g lives in this lane's 16-lane group,
    // so xor-masks {8,4,2,1} reduce exactly over that row.
#pragma unroll
    for (int r = 0; r < 8; ++r) {
      float v0 = s[0][r] * C, v1 = s[1][r] * C;
      float v2 = s[2][r] * C, v3 = s[3][r] * C;
      float rmax = fmaxf(fmaxf(v0, v1), fmaxf(v2, v3));
#pragma unroll
      for (int msk = 8; msk >= 1; msk >>= 1)
        rmax = fmaxf(rmax, __shfl_xor(rmax, msk, 32));
      float mnew = fmaxf(mrow[r], rmax);
      float corr = exp2f(mrow[r] - mnew);
      float p0 = exp2f(v0 - mnew), p1 = exp2f(v1 - mnew);
      float p2 = exp2f(v2 - mnew), p3 = exp2f(v3 - mnew);
      float rsum = (p0 + p1) + (p2 + p3);
#pragma unroll
      for (int msk = 8; msk >= 1; msk >>= 1)
        rsum += __shfl_xor(rsum, msk, 32);
      lrow[r] = lrow[r] * corr + rsum;
      mrow[r] = mnew;
      acc[0][r] *= corr; acc[1][r] *= corr;
      acc[2][r] *= corr; acc[3][r] *= corr;
      // P to LDS (C layout -> row-major 16x64, padded stride)
      int row = (r + 8 * g) * LDP;
      pw[row + nn]      = f2bf_raw(p0);
      pw[row + 16 + nn] = f2bf_raw(p1);
      pw[row + 32 + nn] = f2bf_raw(p2);
      pw[row + 48 + nn] = f2bf_raw(p3);
    }

    // drain own DS stores before cross-lane LDS read (in-order DS pipe)
    asm volatile("s_wait_dscnt 0" ::: "memory");
    v16bf Ap0 = loadA_bf16(pw, LDP, 0,  lane);
    v16bf Ap1 = loadA_bf16(pw, LDP, 32, lane);
    asm volatile("" ::: "memory");

    // O += P(16x64) * V(64x64): B[k][d] = vsh[d][k]
#pragma unroll
    for (int nt = 0; nt < 4; ++nt) {
      const unsigned short* vt = vsh + (nt * 16) * LDP;
      acc[nt] = wmma_bf16(Ap0, loadB_bf16(vt, LDP, 0,  lane), acc[nt]);
      acc[nt] = wmma_bf16(Ap1, loadB_bf16(vt, LDP, 32, lane), acc[nt]);
    }
  }

  // normalize and store ao[(n*SEQ + q)*EMB + h*HD + d] bf16
  unsigned short* obase = ao + (size_t)(n * SEQ + q0) * EMB + h * HD + nn;
#pragma unroll
  for (int r = 0; r < 8; ++r) {
    float inv = 1.0f / lrow[r];
    size_t row = (size_t)(r + 8 * g) * EMB;
    obase[row + 0]  = f2bf_raw(acc[0][r] * inv);
    obase[row + 16] = f2bf_raw(acc[1][r] * inv);
    obase[row + 32] = f2bf_raw(acc[2][r] * inv);
    obase[row + 48] = f2bf_raw(acc[3][r] * inv);
  }
}

// ---------------------------------------------------------------------------
// Kernel 3: out = ao @ Wo^T + bo   (M=8192, N=1024, K=1024), f32 output.
// 1 wave = 64x64 output tile: 16 accumulators, each B tile feeds 4 WMMAs.
// Per K-step: 16 b128 loads -> 16 WMMAs.
// ---------------------------------------------------------------------------
__global__ __launch_bounds__(32) void oproj_kernel(
    const unsigned short* __restrict__ ao, const unsigned short* __restrict__ wob,
    const float* __restrict__ bo, float* __restrict__ out)
{
  int lane = threadIdx.x & 31;
  int m0 = blockIdx.x * 64;
  int n0 = blockIdx.y * 64;
  int g = lane >> 4, nn = lane & 15;

  const unsigned short* ab0 = ao + (size_t)(m0 +  0) * EMB;
  const unsigned short* ab1 = ao + (size_t)(m0 + 16) * EMB;
  const unsigned short* ab2 = ao + (size_t)(m0 + 32) * EMB;
  const unsigned short* ab3 = ao + (size_t)(m0 + 48) * EMB;
  v8f acc[16] = {};

  for (int kb = 0; kb < EMB; kb += 32) {
    __builtin_prefetch(ab0 + (size_t)lane * EMB + kb + 32, 0, 1);
    v16bf A0 = loadA_bf16(ab0, EMB, kb, lane);
    v16bf A1 = loadA_bf16(ab1, EMB, kb, lane);
    v16bf A2 = loadA_bf16(ab2, EMB, kb, lane);
    v16bf A3 = loadA_bf16(ab3, EMB, kb, lane);
#pragma unroll
    for (int nt = 0; nt < 4; ++nt) {
      v16bf B = loadB_bf16(wob + (size_t)(n0 + 16 * nt) * EMB, EMB, kb, lane);
      acc[nt]      = wmma_bf16(A0, B, acc[nt]);
      acc[4 + nt]  = wmma_bf16(A1, B, acc[4 + nt]);
      acc[8 + nt]  = wmma_bf16(A2, B, acc[8 + nt]);
      acc[12 + nt] = wmma_bf16(A3, B, acc[12 + nt]);
    }
  }

  float b[4];
#pragma unroll
  for (int nt = 0; nt < 4; ++nt) b[nt] = bo[n0 + 16 * nt + nn];

#pragma unroll
  for (int mb = 0; mb < 4; ++mb) {
    float* ob = out + (size_t)(m0 + 16 * mb) * EMB + n0 + nn;
#pragma unroll
    for (int r = 0; r < 8; ++r) {
      size_t row = (size_t)(r + 8 * g) * EMB;
#pragma unroll
      for (int nt = 0; nt < 4; ++nt)
        ob[row + 16 * nt] = acc[4 * mb + nt][r] + b[nt];
    }
  }
}

// ---------------------------------------------------------------------------
extern "C" void kernel_launch(void* const* d_in, const int* in_sizes, int n_in,
                              void* d_out, int out_size, void* d_ws, size_t ws_size,
                              hipStream_t stream) {
  const float* values = (const float*)d_in[0];
  const float* keys   = (const float*)d_in[1];
  const float* query  = (const float*)d_in[2];
  const float* Wv     = (const float*)d_in[3];
  const float* Wk     = (const float*)d_in[4];
  const float* Wq     = (const float*)d_in[5];
  const float* Wo     = (const float*)d_in[6];
  const float* bo     = (const float*)d_in[7];
  float* out = (float*)d_out;

  // workspace carve-up (bf16): q' k' v' (16MB each) + attn-out (16MB) + Wo (2MB)
  const size_t QKV = (size_t)NB * HEADS * SEQ * HD;   // 8,388,608
  unsigned short* qp  = (unsigned short*)d_ws;
  unsigned short* kp  = qp + QKV;
  unsigned short* vp  = kp + QKV;
  unsigned short* ao  = vp + QKV;
  unsigned short* wob = ao + (size_t)NB * SEQ * EMB;

  cvt_wo_kernel<<<dim3(EMB), dim3(256), 0, stream>>>(Wo, wob);
  proj_kernel<<<dim3((NB * SEQ) / 16, HEADS, 3), dim3(32), 0, stream>>>(
      query, keys, values, Wq, Wk, Wv, qp, kp, vp);
  attn_kernel<<<dim3(NB * HEADS, SEQ / 64), dim3(128), 0, stream>>>(qp, kp, vp, ao);
  oproj_kernel<<<dim3((NB * SEQ) / 64, EMB / 64), dim3(32), 0, stream>>>(ao, wob, bo, out);
}
